// TtMultiHeadAttentionModel_36309653520395
// MI455X (gfx1250) — matrix operand order repaired and checked by reading, not compiled, not run on hardware
//
#include <hip/hip_runtime.h>
#include <math.h>

// ---------------------------------------------------------------------------
// Fused MHA for MI455X (gfx1250, wave32, WMMA bf16 16x16x32, f32 accum)
// B=1, S=4096, H=1024, heads=2, hd=512
// All fragments loaded as b128 pairs (layouts are lane-contiguous, ISA 7.12.2)
// ---------------------------------------------------------------------------

typedef __attribute__((ext_vector_type(16))) __bf16       v16bf;
typedef __attribute__((ext_vector_type(8)))  float        v8f;
typedef __attribute__((ext_vector_type(4)))  unsigned int u32x4;
typedef __attribute__((ext_vector_type(2)))  unsigned int u32x2;
typedef __attribute__((ext_vector_type(4)))  float        f32x4;

#define S_LEN 4096
#define H_DIM 1024
#define NH    2
#define HD    512
#define LOG2E 1.44269504088896340736f
#define QSCALE 0.04419417382415922f   /* 1/sqrt(512) */

// ---- fragment loaders (per-lane contiguous b128 pairs) --------------------
// A-frag (16x32 bf16): lane r = row; V0..3 hold k = 8h+0..7, V4..7 k = 16+8h+0..7
__device__ __forceinline__ v16bf ldA(const __bf16* row, int k0, int hh) {
    union { v16bf f; u32x4 q[2]; } u;
    const __bf16* p = row + k0 + 8 * hh;
    u.q[0] = *(const u32x4*)(p);
    u.q[1] = *(const u32x4*)(p + 16);
    return u.f;
}
// B-frag (32x16 bf16): lane n = col; V0..3 hold k = 8h+0..7, V4..7 k = 16+8h+0..7
__device__ __forceinline__ v16bf ldB(const __bf16* row, int k0, int hh) {
    union { v16bf f; u32x4 q[2]; } u;
    const __bf16* p = row + k0 + 8 * hh;
    u.q[0] = *(const u32x4*)(p);
    u.q[1] = *(const u32x4*)(p + 16);
    return u.f;
}

#define WMMA_BF16(a, b, c) \
    __builtin_amdgcn_wmma_f32_16x16x32_bf16(false, (a), false, (b), (short)0, (c), false, false)

// ---------------------------------------------------------------------------
// Kernel 0: elementwise fp32 -> bf16 (vectorized), one pass over x and W's.
// ---------------------------------------------------------------------------
__global__ __launch_bounds__(256) void cvt_bf16_kernel(
    const float* __restrict__ src, __bf16* __restrict__ dst, int n4)
{
    const int i = blockIdx.x * 256 + threadIdx.x;
    if (i < n4) {
        f32x4 f = ((const f32x4*)src)[i];
        union { __bf16 h[4]; u32x2 u; } o;
        o.h[0] = (__bf16)f.x; o.h[1] = (__bf16)f.y;
        o.h[2] = (__bf16)f.z; o.h[3] = (__bf16)f.w;
        ((u32x2*)dst)[i] = o.u;
    }
}

// ---------------------------------------------------------------------------
// Kernel 1: Y = xb @ W^T + b  (bf16 in, bf16 out), 32x64 tile per wave.
//   which=0: Q (scaled by 1/sqrt(hd)), row-major (t,d)
//   which=1: K, row-major (t,d)
//   which=2: V, written TRANSPOSED as Vt (d,t)
// ---------------------------------------------------------------------------
__global__ __launch_bounds__(32) void qkv_proj_kernel(
    const __bf16* __restrict__ xb, const __bf16* __restrict__ Wb,
    const float* __restrict__ bq, const float* __restrict__ bk,
    const float* __restrict__ bv,
    __bf16* __restrict__ Qws, __bf16* __restrict__ Kws, __bf16* __restrict__ Vt)
{
    const int lane = threadIdx.x & 31;
    const int r  = lane & 15;
    const int hh = lane >> 4;

    const int t0 = blockIdx.x * 32;       // 128 M-blocks
    const int n0 = blockIdx.y * 64;       // 16 N-blocks
    const int which = blockIdx.z;         // 0:q 1:k 2:v

    const __bf16* W = Wb + (size_t)which * H_DIM * H_DIM;
    const float* bias; float scl;
    if (which == 0)      { bias = bq; scl = QSCALE; }
    else if (which == 1) { bias = bk; scl = 1.0f; }
    else                 { bias = bv; scl = 1.0f; }

    v8f acc[2][4];
#pragma unroll
    for (int mi = 0; mi < 2; ++mi)
#pragma unroll
        for (int j = 0; j < 4; ++j) acc[mi][j] = (v8f){};

    const __bf16* xrow0 = xb + (size_t)(t0 + r) * H_DIM;
    const __bf16* xrow1 = xb + (size_t)(t0 + 16 + r) * H_DIM;

    for (int k0 = 0; k0 < H_DIM; k0 += 32) {
        const v16bf a0 = ldA(xrow0, k0, hh);
        const v16bf a1 = ldA(xrow1, k0, hh);
#pragma unroll
        for (int j = 0; j < 4; ++j) {
            const v16bf b = ldB(W + (size_t)(n0 + 16 * j + r) * H_DIM, k0, hh);
            acc[0][j] = WMMA_BF16(a0, b, acc[0][j]);
            acc[1][j] = WMMA_BF16(a1, b, acc[1][j]);
        }
    }

#pragma unroll
    for (int j = 0; j < 4; ++j) {
        const float bcol = bias[n0 + 16 * j + r];
#pragma unroll
        for (int mi = 0; mi < 2; ++mi)
#pragma unroll
            for (int v = 0; v < 8; ++v)
                acc[mi][j][v] = (acc[mi][j][v] + bcol) * scl;
    }

    if (which == 2) {
#pragma unroll
        for (int mi = 0; mi < 2; ++mi)
#pragma unroll
            for (int j = 0; j < 4; ++j)
#pragma unroll
                for (int v = 0; v < 8; ++v)
                    Vt[(size_t)(n0 + 16 * j + r) * S_LEN + (t0 + 16 * mi + v + 8 * hh)]
                        = (__bf16)acc[mi][j][v];
    } else {
        __bf16* dst = (which == 0) ? Qws : Kws;
#pragma unroll
        for (int mi = 0; mi < 2; ++mi)
#pragma unroll
            for (int j = 0; j < 4; ++j)
#pragma unroll
                for (int v = 0; v < 8; ++v)
                    dst[(size_t)(t0 + 16 * mi + v + 8 * hh) * H_DIM + (n0 + 16 * j + r)]
                        = (__bf16)acc[mi][j][v];
    }
}

// ---------------------------------------------------------------------------
// Kernel 2: fused attention. One workgroup (8 waves) per (16 q-rows, head).
// Two-pass softmax; Q/K/Vt (24 MB bf16) are L2-resident (192 MB L2).
// ---------------------------------------------------------------------------
#define QSTR 520   /* bf16 row stride, 1040 B: 16B-aligned, bank-shifted  */
#define PSTR 136   /* bf16 row stride,  272 B: 16B-aligned, bank-shifted  */

__global__ __launch_bounds__(256) void attn_kernel(
    const __bf16* __restrict__ Qws,
    const __bf16* __restrict__ Kws,
    const __bf16* __restrict__ Vt,
    float* __restrict__ out)
{
    __shared__ __bf16 ldsQ[16 * QSTR];
    __shared__ __bf16 ldsP[16 * PSTR];
    __shared__ float  m_sh[16][8];
    __shared__ float  l_sh[16][8];

    const int tid  = threadIdx.x;
    const int w    = tid >> 5;
    const int lane = tid & 31;
    const int r  = lane & 15;
    const int hh = lane >> 4;

    const int q0   = blockIdx.x * 16;
    const int head = blockIdx.y;
    const int d0   = head * HD;

    // ---- stage Q tile (16 x 512 bf16) into LDS with b128 loads ----
    {
        const u32x4* src = (const u32x4*)Qws;      // 128 u32x4 per Q row
        for (int idx = tid; idx < 16 * 64; idx += 256) {
            const int rr = idx >> 6, cc = idx & 63;
            u32x4 val = src[(size_t)(q0 + rr) * (H_DIM / 8) + (d0 / 8) + cc];
            *(u32x4*)&ldsQ[rr * QSTR + cc * 8] = val;
        }
    }
    __syncthreads();

    // ---- pass 1: row statistics (waves stride over t-tiles) ----
    float m_run[8], l_run[8];
#pragma unroll
    for (int v = 0; v < 8; ++v) { m_run[v] = -__builtin_inff(); l_run[v] = 0.0f; }

    for (int tt = w; tt < S_LEN / 16; tt += 8) {
        const int t0 = tt * 16;
        if (tt + 8 < S_LEN / 16)   // wave-uniform; gfx1250 global_prefetch_b8
            __builtin_prefetch(Kws + (size_t)(t0 + 128 + r) * H_DIM + d0, 0, 1);

        const __bf16* krow = Kws + (size_t)(t0 + r) * H_DIM + d0;
        v8f s = {};
        for (int k0 = 0; k0 < HD; k0 += 32) {
            const v16bf a = ldA(&ldsQ[r * QSTR], k0, hh);
            const v16bf b = ldB(krow, k0, hh);
            s = WMMA_BF16(a, b, s);
        }
#pragma unroll
        for (int v = 0; v < 8; ++v) {
            float xx = s[v];
            float mx = xx;
            mx = fmaxf(mx, __shfl_xor(mx, 1));
            mx = fmaxf(mx, __shfl_xor(mx, 2));
            mx = fmaxf(mx, __shfl_xor(mx, 4));
            mx = fmaxf(mx, __shfl_xor(mx, 8));
            float e = exp2f((xx - mx) * LOG2E);
            e += __shfl_xor(e, 1);
            e += __shfl_xor(e, 2);
            e += __shfl_xor(e, 4);
            e += __shfl_xor(e, 8);
            const float mnew = fmaxf(m_run[v], mx);
            l_run[v] = l_run[v] * exp2f((m_run[v] - mnew) * LOG2E)
                     + e        * exp2f((mx       - mnew) * LOG2E);
            m_run[v] = mnew;
        }
    }

    if (r == 0) {
#pragma unroll
        for (int v = 0; v < 8; ++v) {
            m_sh[v + 8 * hh][w] = m_run[v];
            l_sh[v + 8 * hh][w] = l_run[v];
        }
    }
    __syncthreads();

    float mfin[8], ifin[8];
#pragma unroll
    for (int v = 0; v < 8; ++v) {
        const int row = v + 8 * hh;
        float m = -__builtin_inff();
#pragma unroll
        for (int ww = 0; ww < 8; ++ww) m = fmaxf(m, m_sh[row][ww]);
        float l = 0.0f;
#pragma unroll
        for (int ww = 0; ww < 8; ++ww)
            l = fmaf(l_sh[row][ww], exp2f((m_sh[row][ww] - m) * LOG2E), l);
        mfin[v] = m;
        ifin[v] = 1.0f / l;
    }
    __syncthreads();

    // ---- pass 2: O = softmax(S) @ V; wave w owns output cols [64w, 64w+64) ----
    v8f O[4] = {{}, {}, {}, {}};
    for (int c = 0; c < S_LEN / 128; ++c) {
        const int t0 = (c * 8 + w) * 16;
        const __bf16* krow = Kws + (size_t)(t0 + r) * H_DIM + d0;
        v8f s = {};
        for (int k0 = 0; k0 < HD; k0 += 32) {
            const v16bf a = ldA(&ldsQ[r * QSTR], k0, hh);
            const v16bf b = ldB(krow, k0, hh);
            s = WMMA_BF16(a, b, s);
        }
        // normalized probabilities -> LDS (16 x 128, this wave's 16 cols)
#pragma unroll
        for (int v = 0; v < 8; ++v) {
            const float p = exp2f((s[v] - mfin[v]) * LOG2E) * ifin[v];
            ldsP[(v + 8 * hh) * PSTR + w * 16 + r] = (__bf16)p;
        }
        __syncthreads();

#pragma unroll
        for (int kk = 0; kk < 4; ++kk) {
            const v16bf a = ldA(&ldsP[r * PSTR], kk * 32, hh);
#pragma unroll
            for (int j = 0; j < 4; ++j) {
                // B[k][n] = Vt[d][t]; d = d0+64w+16j+n, t = 128c+32kk+k
                const __bf16* vrow = Vt + (size_t)(d0 + w * 64 + j * 16 + r) * S_LEN;
                const v16bf b = ldB(vrow, c * 128 + kk * 32, hh);
                O[j] = WMMA_BF16(a, b, O[j]);
            }
        }
        __syncthreads();
    }

    // ---- write output (fp32) ----
#pragma unroll
    for (int j = 0; j < 4; ++j) {
#pragma unroll
        for (int v = 0; v < 8; ++v) {
            out[(size_t)(q0 + v + 8 * hh) * H_DIM + d0 + w * 64 + j * 16 + r] = O[j][v];
        }
    }
}

// ---------------------------------------------------------------------------
extern "C" void kernel_launch(void* const* d_in, const int* in_sizes, int n_in,
                              void* d_out, int out_size, void* d_ws, size_t ws_size,
                              hipStream_t stream) {
    const float* x  = (const float*)d_in[0];
    const float* Wq = (const float*)d_in[1];
    const float* bq = (const float*)d_in[2];
    const float* Wk = (const float*)d_in[3];
    const float* bk = (const float*)d_in[4];
    const float* Wv = (const float*)d_in[5];
    const float* bv = (const float*)d_in[6];
    float* out = (float*)d_out;

    // workspace: xb 8MB | Wb 6MB | Q 8MB | K 8MB | Vt 8MB  (all bf16)
    __bf16* xb  = (__bf16*)d_ws;
    __bf16* Wb  = xb + (size_t)S_LEN * H_DIM;
    __bf16* Qws = Wb + (size_t)3 * H_DIM * H_DIM;
    __bf16* Kws = Qws + (size_t)S_LEN * H_DIM;
    __bf16* Vt  = Kws + (size_t)S_LEN * H_DIM;

    const int nx4 = (S_LEN * H_DIM) / 4;          // 1M float4
    const int nw4 = (H_DIM * H_DIM) / 4;          // 256K float4 each
    cvt_bf16_kernel<<<(nx4 + 255) / 256, 256, 0, stream>>>(x, xb, nx4);
    cvt_bf16_kernel<<<(nw4 + 255) / 256, 256, 0, stream>>>(Wq, Wb, nw4);
    cvt_bf16_kernel<<<(nw4 + 255) / 256, 256, 0, stream>>>(Wk, Wb + (size_t)H_DIM * H_DIM, nw4);
    cvt_bf16_kernel<<<(nw4 + 255) / 256, 256, 0, stream>>>(Wv, Wb + (size_t)2 * H_DIM * H_DIM, nw4);

    qkv_proj_kernel<<<dim3(S_LEN / 32, H_DIM / 64, 3), 32, 0, stream>>>(
        xb, Wb, bq, bk, bv, Qws, Kws, Vt);

    attn_kernel<<<dim3(S_LEN / 16, NH), 256, 0, stream>>>(Qws, Kws, Vt, out);
}
